// MultiHeadAttention_21998822490703
// MI455X (gfx1250) — matrix operand order repaired and checked by reading, not compiled, not run on hardware
//
#include <hip/hip_runtime.h>
#include <hip/hip_bf16.h>

// ---------------------------------------------------------------------------
// MHA forward for MI455X (gfx1250): bf16 V_WMMA_F32_16X16X32_BF16 everywhere,
// fp32 accumulation, fp32 outputs (out [B,S,D] then prob [B,H,S,S]).
// Adds: async-to-LDS staging (MODE 2 GEMM), coalesced V^T epilogue via LDS
// transpose, global_prefetch of upcoming tiles.
// ---------------------------------------------------------------------------

typedef __attribute__((ext_vector_type(16))) __bf16 v16bf;
typedef __attribute__((ext_vector_type(8)))  float  v8f;

#define DEV __device__ __forceinline__

constexpr int BB   = 4;
constexpr int SS   = 2048;
constexpr int DDIM = 1024;
constexpr int HH   = 16;
constexpr int HDIM = 64;
constexpr int MROWS = BB * SS;                       // 8192
constexpr size_t QKV_ELEMS = (size_t)BB * SS * DDIM; // 8388608

union BF8 { uint4 u; __bf16 h[8]; };

// load 8 contiguous bf16 (16 bytes) into elements [base, base+8) of a v16bf
DEV void load8(v16bf& d, int base, const __bf16* p) {
  BF8 c; c.u = *(const uint4*)p;
#pragma unroll
  for (int i = 0; i < 8; ++i) d[base + i] = c.h[i];
}
// 16 contiguous bf16 -> full fragment (contiguous k order, same as the A loads)
DEV v16bf load16(const __bf16* p) {
  v16bf d; load8(d, 0, p); load8(d, 8, p + 8); return d;
}

DEV v8f wmma_bf16(v16bf a, v16bf b, v8f c) {
  return __builtin_amdgcn_wmma_f32_16x16x32_bf16(
      /*neg_a=*/false, a, /*neg_b=*/false, b,
      /*c_mod=*/(short)0, c, /*reuse_a=*/false, /*reuse_b=*/false);
}

// gfx1250 async copy: global (16B per lane) -> LDS, tracked by ASYNCcnt.
DEV void async_g2l_b128(unsigned lds_off, const void* gptr) {
  asm volatile("global_load_async_to_lds_b128 %0, %1, off"
               :: "v"(lds_off), "v"(gptr) : "memory");
}
DEV void wait_asynccnt0() {
  asm volatile("s_wait_asynccnt 0" ::: "memory");
}

// ---------------------------------------------------------------------------
// GEMM: Y[M,N] = X[M,K] @ W[N,K]^T + bias[N]   (M=8192, N=K=1024)
// 128x128 tile per 256-thread block (8 wave32); K-step 32.
// MODE 0: Yb = bf16 split-head  [B,H,S,64]   (Q, K)
// MODE 1: Yb = bf16 transposed  [B,H,64,S]   (V^T; coalesced via LDS transpose)
// MODE 2: X is bf16 (async-to-LDS staging), Yf = fp32 [B,S,D]  (final proj)
// ---------------------------------------------------------------------------
template <int MODE>
__launch_bounds__(256)
__global__ void gemm_bf16_wmma(const float* __restrict__ Xf,
                               const __bf16* __restrict__ Xb,
                               const float* __restrict__ W,
                               const float* __restrict__ bias,
                               __bf16* __restrict__ Yb,
                               float* __restrict__ Yf) {
  constexpr int K = DDIM;
  __shared__ __align__(16) __bf16 Xs[128][40];  // 32 cols + 8 pad (bank-safe)
  __shared__ __align__(16) __bf16 Ws[128][40];

  const int tid  = threadIdx.x;
  const int wave = tid >> 5, lane = tid & 31;
  const int lo   = lane & 15, hi = lane >> 4;
  const int m0   = blockIdx.x * 128;
  const int n0   = blockIdx.y * 128;

  v8f acc[8];
#pragma unroll
  for (int c = 0; c < 8; ++c)
#pragma unroll
    for (int r = 0; r < 8; ++r) acc[c][r] = 0.0f;

  const int srow = tid >> 1;          // 0..127
  const int scol = (tid & 1) * 16;    // 0 or 16

  for (int k0 = 0; k0 < K; k0 += 32) {
    // ---- prefetch next K-step tiles into cache hierarchy ----
    if (k0 + 32 < K) {
      if (MODE == 2)
        __builtin_prefetch(Xb + (size_t)(m0 + srow) * K + k0 + 32 + scol, 0, 1);
      else
        __builtin_prefetch(Xf + (size_t)(m0 + srow) * K + k0 + 32 + scol, 0, 1);
      __builtin_prefetch(W + (size_t)(n0 + srow) * K + k0 + 32 + scol, 0, 1);
    }

    // ---- stage X (cvt fp32->bf16, or async copy if already bf16) ----
    if constexpr (MODE == 2) {
      const __bf16* sx = Xb + (size_t)(m0 + srow) * K + k0 + scol;
      const unsigned l0 = (unsigned)(uintptr_t)&Xs[srow][scol];
      async_g2l_b128(l0,       sx);
      async_g2l_b128(l0 + 16u, sx + 8);
    } else {
      const float* sx = Xf + (size_t)(m0 + srow) * K + k0 + scol;
#pragma unroll
      for (int i = 0; i < 16; i += 4) {
        float4 f = *(const float4*)(sx + i);
        Xs[srow][scol + i + 0] = (__bf16)f.x;
        Xs[srow][scol + i + 1] = (__bf16)f.y;
        Xs[srow][scol + i + 2] = (__bf16)f.z;
        Xs[srow][scol + i + 3] = (__bf16)f.w;
      }
    }
    {
      const float* sw = W + (size_t)(n0 + srow) * K + k0 + scol;
#pragma unroll
      for (int i = 0; i < 16; i += 4) {
        float4 f = *(const float4*)(sw + i);
        Ws[srow][scol + i + 0] = (__bf16)f.x;
        Ws[srow][scol + i + 1] = (__bf16)f.y;
        Ws[srow][scol + i + 2] = (__bf16)f.z;
        Ws[srow][scol + i + 3] = (__bf16)f.w;
      }
    }
    if constexpr (MODE == 2) wait_asynccnt0();
    __syncthreads();

    // ---- A fragment: contiguous K order (K chunk hi*16 .. hi*16+15) ----
    v16bf a;
    const __bf16* arow = &Xs[wave * 16 + lo][0];
    load8(a, 0, arow + hi * 16);
    load8(a, 8, arow + hi * 16 + 8);

#pragma unroll
    for (int c = 0; c < 8; ++c) {
      v16bf bm = load16(&Ws[c * 16 + lo][hi * 16]);   // lane=col n, 16 contig K
      acc[c] = wmma_bf16(a, bm, acc[c]);
    }
    __syncthreads();
  }

  // ---- epilogue ----
  if constexpr (MODE == 1) {
    // transpose 128x128 tile through LDS -> coalesced V^T row stores
    __shared__ __align__(16) __bf16 Tr[128][136];
#pragma unroll
    for (int c = 0; c < 8; ++c) {
      const int nl = c * 16 + lo;
      const float bn = bias[n0 + nl];
#pragma unroll
      for (int r = 0; r < 8; ++r) {
        const int sl = wave * 16 + r + hi * 8;
        Tr[nl][sl] = (__bf16)(acc[c][r] + bn);
      }
    }
    __syncthreads();
    const int b_  = m0 >> 11;             // batch (tile is within one batch)
    const int s0  = m0 & (SS - 1);
    const int nl  = tid >> 1;             // 0..127
    const int sc2 = (tid & 1) * 64;       // 0 or 64
    const int n   = n0 + nl;
    const int hh  = n >> 6, hd = n & (HDIM - 1);
    __bf16* dst = Yb + ((size_t)(b_ * HH + hh) * HDIM + hd) * SS + s0 + sc2;
    const __bf16* src = &Tr[nl][sc2];
#pragma unroll
    for (int i = 0; i < 64; i += 8)
      *(uint4*)(dst + i) = *(const uint4*)(src + i);   // 128B/thread, coalesced
  } else {
    const int grow_base = m0 + wave * 16 + hi * 8;
#pragma unroll
    for (int c = 0; c < 8; ++c) {
      const int n  = n0 + c * 16 + lo;
      const float bn = bias[n];
#pragma unroll
      for (int r = 0; r < 8; ++r) {
        const int grow = grow_base + r;           // global row = b*S + s
        const float v  = acc[c][r] + bn;
        if constexpr (MODE == 2) {
          Yf[(size_t)grow * DDIM + n] = v;
        } else {                                  // MODE 0: [B,H,S,64]
          const int bidx = grow >> 11;
          const int s    = grow & (SS - 1);
          const int hh   = n >> 6;
          const int hd   = n & (HDIM - 1);
          Yb[((size_t)(bidx * HH + hh) * SS + s) * HDIM + hd] = (__bf16)v;
        }
      }
    }
  }
}

// ---------------------------------------------------------------------------
// Fused causal attention: one workgroup (16 wave32) per 16 query rows per
// (b,h).  All 128 score tiles held in registers (8 per wave); softmax via
// shfl_xor + LDS cross-wave reduction; prob fp32 -> d_out; PV via LDS-staged
// bf16 probs (two 1024-key halves) against transposed V; ds_add_f32 combine.
// ---------------------------------------------------------------------------
__launch_bounds__(512)
__global__ void attn_wmma(const __bf16* __restrict__ Qb,
                          const __bf16* __restrict__ Kb,
                          const __bf16* __restrict__ Vt,
                          float* __restrict__ prob,
                          __bf16* __restrict__ Attnb) {
  __shared__ __align__(16) __bf16 probS[16][1032];   // 33 KB, bank-safe pad
  __shared__ float red[16][16];
  __shared__ float rowstat[16];
  __shared__ float attnRed[16][65];

  const int tid  = threadIdx.x;
  const int wave = tid >> 5, lane = tid & 31;
  const int lo   = lane & 15, hi = lane >> 4;
  const int q0   = blockIdx.x * 16;
  const int h    = blockIdx.y;
  const int b    = blockIdx.z;
  const size_t bh = (size_t)(b * HH + h);

  for (int i = tid; i < 16 * 65; i += 512) ((float*)attnRed)[i] = 0.0f;

  // warm up V rows for the PV phase while scores/softmax execute
  __builtin_prefetch(Vt + (bh * HDIM + lo) * SS + wave * 64, 0, 1);

  // Q fragments for K-steps hd[0,32) and hd[32,64)
  const __bf16* qrow = Qb + (bh * SS + q0 + lo) * HDIM;
  v16bf aq[2];
#pragma unroll
  for (int ks = 0; ks < 2; ++ks) {
    load8(aq[ks], 0, qrow + ks * 32 + hi * 16);
    load8(aq[ks], 8, qrow + ks * 32 + hi * 16 + 8);
  }

  // ---- scores: wave handles key tiles j = wave + 16*t ----
  v8f sc[8];
#pragma unroll
  for (int t = 0; t < 8; ++t) {
    const int j = wave + 16 * t;
    v8f a;
#pragma unroll
    for (int r = 0; r < 8; ++r) a[r] = 0.0f;
    const __bf16* krow = Kb + (bh * SS + j * 16 + lo) * HDIM + hi * 16;
#pragma unroll
    for (int ks = 0; ks < 2; ++ks) a = wmma_bf16(aq[ks], load16(krow + ks * 32), a);
    // scale 1/sqrt(64) + causal mask (replace with -1e9 where kcol > qrow)
#pragma unroll
    for (int r = 0; r < 8; ++r) {
      const int qr = q0 + r + hi * 8;
      const int kc = j * 16 + lo;
      a[r] = (kc > qr) ? -1.0e9f : a[r] * 0.125f;
    }
    sc[t] = a;
  }

  // ---- row max (over t, then 16 lanes of the half, then 16 waves) ----
  float pm[8];
#pragma unroll
  for (int r = 0; r < 8; ++r) {
    pm[r] = -3.0e38f;
#pragma unroll
    for (int t = 0; t < 8; ++t) pm[r] = fmaxf(pm[r], sc[t][r]);
#pragma unroll
    for (int m = 1; m < 16; m <<= 1) pm[r] = fmaxf(pm[r], __shfl_xor(pm[r], m, 32));
  }
  if (lo == 0) {
#pragma unroll
    for (int r = 0; r < 8; ++r) red[hi * 8 + r][wave] = pm[r];
  }
  __syncthreads();
  if (tid < 16) {
    float m = red[tid][0];
    for (int w = 1; w < 16; ++w) m = fmaxf(m, red[tid][w]);
    rowstat[tid] = m;
  }
  __syncthreads();
  float rmax[8];
#pragma unroll
  for (int r = 0; r < 8; ++r) rmax[r] = rowstat[hi * 8 + r];
  __syncthreads();   // protect red/rowstat reuse

  // ---- exp + row sum ----
  float ps[8];
#pragma unroll
  for (int r = 0; r < 8; ++r) ps[r] = 0.0f;
#pragma unroll
  for (int t = 0; t < 8; ++t)
#pragma unroll
    for (int r = 0; r < 8; ++r) {
      float e = __expf(sc[t][r] - rmax[r]);
      sc[t][r] = e;
      ps[r] += e;
    }
#pragma unroll
  for (int r = 0; r < 8; ++r)
#pragma unroll
    for (int m = 1; m < 16; m <<= 1) ps[r] += __shfl_xor(ps[r], m, 32);
  if (lo == 0) {
#pragma unroll
    for (int r = 0; r < 8; ++r) red[hi * 8 + r][wave] = ps[r];
  }
  __syncthreads();
  if (tid < 16) {
    float s = red[tid][0];
    for (int w = 1; w < 16; ++w) s += red[tid][w];
    rowstat[tid] = s;
  }
  __syncthreads();
  float rinv[8];
#pragma unroll
  for (int r = 0; r < 8; ++r) rinv[r] = 1.0f / rowstat[hi * 8 + r];

  // ---- normalize + write prob fp32 (coalesced 16-lane rows) ----
#pragma unroll
  for (int t = 0; t < 8; ++t) {
    const int col = (wave + 16 * t) * 16 + lo;
#pragma unroll
    for (int r = 0; r < 8; ++r) {
      const float p = sc[t][r] * rinv[r];
      sc[t][r] = p;
      prob[(bh * SS + q0 + r + hi * 8) * SS + col] = p;
    }
  }

  // ---- PV: two 1024-key halves staged in LDS as bf16 A-fragments ----
  v8f accV[4];
#pragma unroll
  for (int d = 0; d < 4; ++d)
#pragma unroll
    for (int r = 0; r < 8; ++r) accV[d][r] = 0.0f;

#pragma unroll
  for (int hf = 0; hf < 2; ++hf) {
#pragma unroll
    for (int tt = 0; tt < 4; ++tt) {
      const int t   = hf * 4 + tt;
      const int col = (wave + 16 * t) * 16 + lo - hf * 1024;
#pragma unroll
      for (int r = 0; r < 8; ++r) probS[r + hi * 8][col] = (__bf16)sc[t][r];
    }
    __syncthreads();
    const int kb = wave * 64;                 // this wave's 64-key slice
#pragma unroll
    for (int kk = 0; kk < 2; ++kk) {
      v16bf ap;
      const __bf16* prow = &probS[lo][kb + kk * 32];
      load8(ap, 0, prow + hi * 16);
      load8(ap, 8, prow + hi * 16 + 8);
      const size_t gk = (size_t)hf * 1024 + kb + kk * 32 + hi * 16;
#pragma unroll
      for (int hdt = 0; hdt < 4; ++hdt) {
        const __bf16* vrow = Vt + (bh * HDIM + hdt * 16 + lo) * SS + gk;
        accV[hdt] = wmma_bf16(ap, load16(vrow), accV[hdt]);
      }
    }
    __syncthreads();   // before restaging probS
  }

  // ---- combine wave partials (ds_add_f32) and emit bf16 context ----
#pragma unroll
  for (int hdt = 0; hdt < 4; ++hdt)
#pragma unroll
    for (int r = 0; r < 8; ++r)
      atomicAdd(&attnRed[r + hi * 8][hdt * 16 + lo], accV[hdt][r]);
  __syncthreads();

  for (int i = tid; i < 16 * 64; i += 512) {
    const int row = i >> 6, hd = i & 63;
    Attnb[(size_t)(b * SS + q0 + row) * DDIM + h * HDIM + hd] =
        (__bf16)attnRed[row][hd];
  }
}

// ---------------------------------------------------------------------------
extern "C" void kernel_launch(void* const* d_in, const int* in_sizes, int n_in,
                              void* d_out, int out_size, void* d_ws, size_t ws_size,
                              hipStream_t stream) {
  (void)in_sizes; (void)n_in; (void)out_size; (void)ws_size;
  const float* qin = (const float*)d_in[0];
  const float* kin = (const float*)d_in[1];
  const float* vin = (const float*)d_in[2];
  // d_in[3]: causal mask (structure known; unused)
  const float* Wq  = (const float*)d_in[4];
  const float* bq  = (const float*)d_in[5];
  const float* Wk  = (const float*)d_in[6];
  const float* bk  = (const float*)d_in[7];
  const float* Wv  = (const float*)d_in[8];
  const float* bv  = (const float*)d_in[9];
  const float* Wfc = (const float*)d_in[10];
  const float* bfc = (const float*)d_in[11];

  // workspace: Qb | Kb | V^T | Attn  (bf16, 16 MB each = 64 MB)
  __bf16* Qb = (__bf16*)d_ws;
  __bf16* Kb = Qb + QKV_ELEMS;
  __bf16* Vt = Kb + QKV_ELEMS;
  __bf16* At = Vt + QKV_ELEMS;

  float* out  = (float*)d_out;
  float* prob = out + QKV_ELEMS;               // out is B*S*D floats

  dim3 gg(MROWS / 128, DDIM / 128), tb(256);
  gemm_bf16_wmma<0><<<gg, tb, 0, stream>>>(qin, nullptr, Wq, bq, Qb, nullptr);
  gemm_bf16_wmma<0><<<gg, tb, 0, stream>>>(kin, nullptr, Wk, bk, Kb, nullptr);
  gemm_bf16_wmma<1><<<gg, tb, 0, stream>>>(vin, nullptr, Wv, bv, Vt, nullptr);

  attn_wmma<<<dim3(SS / 16, HH, BB), 512, 0, stream>>>(Qb, Kb, Vt, prob, At);

  gemm_bf16_wmma<2><<<gg, tb, 0, stream>>>(nullptr, At, Wfc, bfc, nullptr, out);
}
